// SSDiM_41781441856314
// MI455X (gfx1250) — compile-verified, matrix operand order checked by reading, not run on hardware
//
#include <hip/hip_runtime.h>
#include <hip/hip_bf16.h>

// ---------------------------------------------------------------------------
// SSDiM block for gfx1250 (MI455X). All GEMMs via v_wmma_f32_16x16x32_f16,
// B-panels staged in LDS with GLOBAL_LOAD_ASYNC_TO_LDS_B128 (ASYNCcnt).
// ---------------------------------------------------------------------------

typedef _Float16 f16;
typedef __attribute__((ext_vector_type(16))) _Float16 v16h;
typedef __attribute__((ext_vector_type(8)))  float    v8f;

#define NGRID 16
#define LSEQ  256      // L
#define BB    2        // B
#define DD    512      // D
#define DS_   16
#define DC_   4
#define DTR_  32
#define PP    8
#define NSEQ  (PP*BB)          // 16 sequences
#define MTOT  (NSEQ*LSEQ)      // 4096 rows for mamba GEMMs
#define MBL   (BB*LSEQ)        // 512 rows for MLP GEMMs

// ---- CDNA5 async global->LDS copy (ASYNCcnt) with safe fallback -----------
#if defined(__has_builtin)
#if __has_builtin(__builtin_amdgcn_global_load_async_to_lds_b128)
#define HAVE_ASYNC_LDS 1
#endif
#endif
#ifndef HAVE_ASYNC_LDS
#define HAVE_ASYNC_LDS 0
#endif

typedef int vi4 __attribute__((vector_size(16)));
typedef vi4 __attribute__((address_space(1))) *g128_t;   // global v4i32*
typedef vi4 __attribute__((address_space(3))) *l128_t;   // LDS    v4i32*

__device__ inline void copy16_g2lds(const f16* g, f16* l) {
#if HAVE_ASYNC_LDS
    __builtin_amdgcn_global_load_async_to_lds_b128((g128_t)g, (l128_t)l, 0, 0);
#else
    *(uint4*)l = *(const uint4*)g;   // global_load_b128 + ds_store_b128
#endif
}

__device__ inline void wait_async_copies() {
#if HAVE_ASYNC_LDS
#if __has_builtin(__builtin_amdgcn_s_wait_asynccnt)
    __builtin_amdgcn_s_wait_asynccnt(0);
#else
    asm volatile("s_wait_asynccnt 0" ::: "memory");
#endif
#endif
}

// ---------------------------------------------------------------------------
// Fragment loader: 16-bit A-matrix 16x32 (MxK) layout (ISA 7.12.2):
//   lanes 0-15  : M=lane,    elems 0..7 -> K = k0+0..7,   elems 8..15 -> K = k0+16..23
//   lanes 16-31 : M=lane-16, elems 0..7 -> K = k0+8..15,  elems 8..15 -> K = k0+24..31
// B is pre-transposed to NxK so the same loader builds B-fragments (N as "row").
// Works on global or LDS-backed pointers (8-half contiguous runs -> b128).
// ---------------------------------------------------------------------------
__device__ inline v16h load_frag16(const f16* A, int ld, int row0, int k0, int lane) {
    const int hl = lane & 15;
    const int hi = (lane >> 4) & 1;
    const f16* base = A + (size_t)(row0 + hl) * (size_t)ld + k0;
    const int o0 = hi ? 8 : 0;
    const int o1 = hi ? 24 : 16;
    v16h a;
#pragma unroll
    for (int e = 0; e < 8; ++e) a[e]     = base[o0 + e];
#pragma unroll
    for (int e = 0; e < 8; ++e) a[8 + e] = base[o1 + e];
    return a;
}

__device__ inline float sigmoidf_(float v) { return 1.0f / (1.0f + __expf(-v)); }
__device__ inline float siluf_(float v)    { return v * sigmoidf_(v); }
__device__ inline float softplusf_(float v){ return (v > 20.0f) ? v : log1pf(__expf(v)); }
__device__ inline float gelu_tanh_(float v){
    float c = 0.7978845608028654f * (v + 0.044715f * v * v * v);
    return 0.5f * v * (1.0f + tanhf(c));
}

// ---------------------------------------------------------------------------
// WMMA GEMM: C[M,N](f32) = A[M,K](f16, lda) * Bt[N,K](f16) (+epilogue)
// modes: 0 = f32 store
//        1 = +bias, f32 store
//        2 = +bias, softplus, f32 store          (dt path)
//        3 = +bias, tanh-GELU, f16 store to Ch   (MLP mid)
//        4 = f32 store to C AND f16 store to Ch  (x_dbl)
// block = (32,8): 8 waves; block computes 128(M) x 64(N); B panel (64x32 f16)
// double-buffered in LDS, staged with async global->LDS copies.
// REQUIREMENT: M % 128 == 0, N % 64 == 0, K % 32 == 0 (true for all calls).
// ---------------------------------------------------------------------------
__global__ __launch_bounds__(256)
void wmma_gemm_kernel(const f16* __restrict__ A, const f16* __restrict__ Bt,
                      const float* __restrict__ bias,
                      float* __restrict__ C, f16* __restrict__ Ch,
                      int M, int N, int K, int lda, int mode) {
    __shared__ f16 Bs[2][64 * 32];       // 2 x 4KB double buffer

    const int lane = threadIdx.x;        // 0..31
    const int wave = threadIdx.y;        // 0..7
    const int tid  = wave * 32 + lane;   // 0..255
    const int row0 = blockIdx.y * 128 + wave * 16;
    const int colb = blockIdx.x * 64;

    // staging role: each of 256 threads copies 16B; panel = 64 rows x 32 k
    const int sn = tid >> 2;             // 0..63  (panel row = column of C)
    const int sk = (tid & 3) * 8;        // 0,8,16,24

    v8f acc0 = {}, acc1 = {}, acc2 = {}, acc3 = {};

    const int nk = K / 32;
    // prologue: stage panel 0
    copy16_g2lds(Bt + (size_t)(colb + sn) * K + 0 + sk, &Bs[0][sn * 32 + sk]);

    for (int it = 0; it < nk; ++it) {
        const int k0  = it * 32;
        const int cur = it & 1;
        wait_async_copies();
        __syncthreads();                 // panel 'cur' visible to all waves
        if (it + 1 < nk) {               // stage next panel into other buffer
            copy16_g2lds(Bt + (size_t)(colb + sn) * K + (k0 + 32) + sk,
                         &Bs[cur ^ 1][sn * 32 + sk]);
        }

        v16h a = load_frag16(A, lda, row0, k0, lane);
        if (it + 1 < nk) {
            // CDNA5 speculative prefetch of next A K-slice (global_prefetch_b8)
            __builtin_prefetch(A + (size_t)(row0 + (lane & 15)) * (size_t)lda + k0 + 32, 0, 1);
        }
        {
            v16h b = load_frag16(&Bs[cur][0], 32,  0, 0, lane);
            acc0 = __builtin_amdgcn_wmma_f32_16x16x32_f16(false, a, false, b, (short)0, acc0, false, false);
        }
        {
            v16h b = load_frag16(&Bs[cur][0], 32, 16, 0, lane);
            acc1 = __builtin_amdgcn_wmma_f32_16x16x32_f16(false, a, false, b, (short)0, acc1, false, false);
        }
        {
            v16h b = load_frag16(&Bs[cur][0], 32, 32, 0, lane);
            acc2 = __builtin_amdgcn_wmma_f32_16x16x32_f16(false, a, false, b, (short)0, acc2, false, false);
        }
        {
            v16h b = load_frag16(&Bs[cur][0], 32, 48, 0, lane);
            acc3 = __builtin_amdgcn_wmma_f32_16x16x32_f16(false, a, false, b, (short)0, acc3, false, false);
        }
        __syncthreads();                 // all waves done reading 'cur' before it
                                         // is overwritten in the next iteration
    }

    // C/D layout (ISA 7.12.2): lanes 0-15: (M=row0+i, N=col+lane);
    //                          lanes 16-31: (M=row0+8+i, N=col+lane-16)
    const int nlane = lane & 15;
    const int mbase = row0 + (((lane >> 4) & 1) << 3);
    v8f accs[4] = { acc0, acc1, acc2, acc3 };
#pragma unroll
    for (int j = 0; j < 4; ++j) {
        const int nn = colb + j * 16 + nlane;
        if (nn >= N) continue;
        const float bval = bias ? bias[nn] : 0.0f;
#pragma unroll
        for (int i = 0; i < 8; ++i) {
            const int mm = mbase + i;
            float v = accs[j][i];
            const size_t oidx = (size_t)mm * (size_t)N + nn;
            if (mode == 3) {
                Ch[oidx] = (f16)gelu_tanh_(v + bval);
            } else if (mode == 2) {
                C[oidx] = softplusf_(v + bval);
            } else if (mode == 1) {
                C[oidx] = v + bval;
            } else if (mode == 4) {
                C[oidx]  = v;
                Ch[oidx] = (f16)v;
            } else {
                C[oidx] = v;
            }
        }
    }
}

// ---------------------------------------------------------------------------
// Weight convert: W[K,N] f32 row-major -> Wt[N,K] f16 (transposed for B frags)
// ---------------------------------------------------------------------------
__global__ void transpose_f16_kernel(const float* __restrict__ W, f16* __restrict__ Wt,
                                     int K, int N) {
    int idx = blockIdx.x * blockDim.x + threadIdx.x;
    if (idx >= K * N) return;
    int k = idx / N, n = idx % N;
    Wt[(size_t)n * K + k] = (f16)W[idx];
}

// ---------------------------------------------------------------------------
// LayerNorm + modulate (+ optional path-gather). 128 threads, 1 block per row.
// gather==1: blockIdx.x = (p*B+b)*L + l, src row = (b, paths[p,l]), t offs 0/D
// gather==0: blockIdx.x = b*L + l, t offs 3D/4D
// ---------------------------------------------------------------------------
__global__ __launch_bounds__(128)
void ln_mod_kernel(const float* __restrict__ xin, const float* __restrict__ t,
                   const int* __restrict__ paths, f16* __restrict__ outp, int gather) {
    __shared__ float sS[128], sQ[128];
    const int bid = blockIdx.x;
    const int tid = threadIdx.x;
    int b, srcRow, sh_off, sc_off;
    if (gather) {
        int l  = bid % LSEQ;
        int pb = bid / LSEQ;
        b = pb % BB;
        int p = pb / BB;
        srcRow = b * LSEQ + paths[p * LSEQ + l];
        sh_off = 0; sc_off = DD;
    } else {
        b = bid / LSEQ;
        srcRow = bid;
        sh_off = 3 * DD; sc_off = 4 * DD;
    }
    const float* row = xin + (size_t)srcRow * DD;

    float s = 0.0f, q = 0.0f;
#pragma unroll
    for (int j = 0; j < 4; ++j) {
        float v = row[tid + 128 * j];
        s += v; q += v * v;
    }
    sS[tid] = s; sQ[tid] = q;
    __syncthreads();
    for (int st = 64; st > 0; st >>= 1) {
        if (tid < st) { sS[tid] += sS[tid + st]; sQ[tid] += sQ[tid + st]; }
        __syncthreads();
    }
    const float mean = sS[0] * (1.0f / DD);
    const float var  = sQ[0] * (1.0f / DD) - mean * mean;
    const float inv  = rsqrtf(var + 1e-6f);

    f16* out = outp + (size_t)bid * DD;
    const float* tb = t + (size_t)b * 6 * DD;
#pragma unroll
    for (int j = 0; j < 4; ++j) {
        int d = tid + 128 * j;
        float xn = (row[d] - mean) * inv;
        float m  = xn * (1.0f + tb[sc_off + d]) + tb[sh_off + d];
        out[d] = (f16)m;
    }
}

// ---------------------------------------------------------------------------
// Depthwise causal conv(DC=4) + bias + SiLU. xm = xz[:, :, 0:512] (stride 1024)
// writes xc (f32) and xc_h (f16).
// ---------------------------------------------------------------------------
__global__ void conv_silu_kernel(const float* __restrict__ xz,
                                 const float* __restrict__ conv_w,
                                 const float* __restrict__ conv_b,
                                 float* __restrict__ xc, f16* __restrict__ xch) {
    int idx = blockIdx.x * blockDim.x + threadIdx.x;
    if (idx >= NSEQ * LSEQ * DD) return;
    int d = idx % DD;
    int l = (idx / DD) % LSEQ;
    int n = idx / (DD * LSEQ);
    float acc = conv_b[d];
#pragma unroll
    for (int k = 0; k < DC_; ++k) {
        int lp = l - (DC_ - 1) + k;
        if (lp >= 0) acc += conv_w[d * DC_ + k] * xz[((size_t)(n * LSEQ + lp)) * (2 * DD) + d];
    }
    float v = siluf_(acc);
    xc[idx]  = v;
    xch[idx] = (f16)v;
}

// ---------------------------------------------------------------------------
// Selective scan: one block per sequence (16 blocks), 512 threads = channels.
// h[16] states in registers; B/C for the timestep staged in LDS.
// Fused: ym = (y + D_param*xc) * silu(z)  ->  f16 for the W_out GEMM.
// ---------------------------------------------------------------------------
__global__ __launch_bounds__(512)
void ssm_scan_kernel(const float* __restrict__ dt, const float* __restrict__ xdbl,
                     const float* __restrict__ xz, const float* __restrict__ xc,
                     const float* __restrict__ A_log, const float* __restrict__ D_param,
                     f16* __restrict__ ym_h) {
    __shared__ float bc[2 * DS_];          // B (16) then C (16)
    const int n = blockIdx.x;
    const int d = threadIdx.x;

    float A_d[DS_];
#pragma unroll
    for (int k = 0; k < DS_; ++k) A_d[k] = -__expf(A_log[d * DS_ + k]);

    float h[DS_];
#pragma unroll
    for (int k = 0; k < DS_; ++k) h[k] = 0.0f;

    const float Dp = D_param[d];

    for (int l = 0; l < LSEQ; ++l) {
        if (d < 2 * DS_) bc[d] = xdbl[((size_t)(n * LSEQ + l)) * (DTR_ + 2 * DS_) + DTR_ + d];
        __syncthreads();
        const size_t ridx = (size_t)(n * LSEQ + l);
        const float dtv = dt[ridx * DD + d];
        const float xcv = xc[ridx * DD + d];
        float y = 0.0f;
#pragma unroll
        for (int k = 0; k < DS_; ++k) {
            float dA = __expf(dtv * A_d[k]);
            h[k] = dA * h[k] + dtv * bc[k] * xcv;
            y += h[k] * bc[DS_ + k];
        }
        const float zv = xz[ridx * (2 * DD) + DD + d];
        float ym = (y + Dp * xcv) * siluf_(zv);
        ym_h[ridx * DD + d] = (f16)ym;
        __syncthreads();
    }
}

// ---------------------------------------------------------------------------
// Un-permute (paths_rev), mean over P, x1 = x + g_msa*mean + y
// ---------------------------------------------------------------------------
__global__ void mean_residual_kernel(const float* __restrict__ mo,
                                     const int* __restrict__ paths_rev,
                                     const float* __restrict__ x, const float* __restrict__ y,
                                     const float* __restrict__ t, float* __restrict__ x1) {
    int idx = blockIdx.x * blockDim.x + threadIdx.x;
    if (idx >= BB * LSEQ * DD) return;
    int d = idx % DD;
    int l = (idx / DD) % LSEQ;
    int b = idx / (DD * LSEQ);
    float acc = 0.0f;
#pragma unroll
    for (int p = 0; p < PP; ++p) {
        int lr = paths_rev[p * LSEQ + l];
        acc += mo[((size_t)((p * BB + b) * LSEQ + lr)) * DD + d];
    }
    float g = t[(size_t)b * 6 * DD + 2 * DD + d];
    x1[idx] = x[idx] + g * (acc * (1.0f / PP)) + y[idx];
}

// ---------------------------------------------------------------------------
// out = x1 + g_mlp * mlp2
// ---------------------------------------------------------------------------
__global__ void final_residual_kernel(const float* __restrict__ x1,
                                      const float* __restrict__ mlp2,
                                      const float* __restrict__ t, float* __restrict__ out) {
    int idx = blockIdx.x * blockDim.x + threadIdx.x;
    if (idx >= BB * LSEQ * DD) return;
    int d = idx % DD;
    int b = idx / (DD * LSEQ);
    float g = t[(size_t)b * 6 * DD + 5 * DD + d];
    out[idx] = x1[idx] + g * mlp2[idx];
}

// ---------------------------------------------------------------------------
// Host-side orchestration
// ---------------------------------------------------------------------------
extern "C" void kernel_launch(void* const* d_in, const int* in_sizes, int n_in,
                              void* d_out, int out_size, void* d_ws, size_t ws_size,
                              hipStream_t stream) {
    const float* x       = (const float*)d_in[0];
    const float* t       = (const float*)d_in[1];
    const float* y       = (const float*)d_in[2];
    const int*   paths   = (const int*)  d_in[3];
    const int*   prev    = (const int*)  d_in[4];
    const float* W_in    = (const float*)d_in[5];
    const float* conv_w  = (const float*)d_in[6];
    const float* conv_b  = (const float*)d_in[7];
    const float* W_xproj = (const float*)d_in[8];
    const float* W_dt    = (const float*)d_in[9];
    const float* b_dt    = (const float*)d_in[10];
    const float* A_log   = (const float*)d_in[11];
    const float* D_param = (const float*)d_in[12];
    const float* W_out   = (const float*)d_in[13];
    const float* W_mlp1  = (const float*)d_in[14];
    const float* b_mlp1  = (const float*)d_in[15];
    const float* W_mlp2  = (const float*)d_in[16];
    const float* b_mlp2  = (const float*)d_in[17];
    float* out = (float*)d_out;

    char* base = (char*)d_ws;
    size_t off = 0;
    auto alloc = [&](size_t bytes) -> char* {
        off = (off + 255) & ~(size_t)255;
        char* p = base + off;
        off += bytes;
        return p;
    };

    f16*   tp_h     = (f16*)  alloc((size_t)MTOT * DD * 2);
    f16*   Wt_in    = (f16*)  alloc((size_t)(2 * DD) * DD * 2);
    f16*   Wt_xproj = (f16*)  alloc((size_t)(DTR_ + 2 * DS_) * DD * 2);
    f16*   Wt_dt    = (f16*)  alloc((size_t)DD * DTR_ * 2);
    f16*   Wt_out   = (f16*)  alloc((size_t)DD * DD * 2);
    f16*   Wt_mlp1  = (f16*)  alloc((size_t)(4 * DD) * DD * 2);
    f16*   Wt_mlp2  = (f16*)  alloc((size_t)DD * (4 * DD) * 2);
    float* xz       = (float*)alloc((size_t)MTOT * (2 * DD) * 4);
    float* xc       = (float*)alloc((size_t)MTOT * DD * 4);
    f16*   xc_h     = (f16*)  alloc((size_t)MTOT * DD * 2);
    float* x_dbl    = (float*)alloc((size_t)MTOT * (DTR_ + 2 * DS_) * 4);
    f16*   x_dbl_h  = (f16*)  alloc((size_t)MTOT * (DTR_ + 2 * DS_) * 2);
    float* dt_buf   = (float*)alloc((size_t)MTOT * DD * 4);
    f16*   ym_h     = (f16*)  alloc((size_t)MTOT * DD * 2);
    float* mo       = (float*)alloc((size_t)MTOT * DD * 4);
    float* x1       = (float*)alloc((size_t)MBL * DD * 4);
    f16*   hmod_h   = (f16*)  alloc((size_t)MBL * DD * 2);
    f16*   mlp1_h   = (f16*)  alloc((size_t)MBL * (4 * DD) * 2);
    float* mlp2     = (float*)alloc((size_t)MBL * DD * 4);
    (void)ws_size; (void)in_sizes; (void)n_in; (void)out_size;

    const int T = 256;
    // ---- weight conversions (f32 KxN -> f16 NxK) ----
    transpose_f16_kernel<<<(DD * 2 * DD + T - 1) / T, T, 0, stream>>>(W_in,    Wt_in,    DD,     2 * DD);
    transpose_f16_kernel<<<(DD * 64 + T - 1) / T,     T, 0, stream>>>(W_xproj, Wt_xproj, DD,     64);
    transpose_f16_kernel<<<(DTR_ * DD + T - 1) / T,   T, 0, stream>>>(W_dt,    Wt_dt,    DTR_,   DD);
    transpose_f16_kernel<<<(DD * DD + T - 1) / T,     T, 0, stream>>>(W_out,   Wt_out,   DD,     DD);
    transpose_f16_kernel<<<(DD * 4 * DD + T - 1) / T, T, 0, stream>>>(W_mlp1,  Wt_mlp1,  DD,     4 * DD);
    transpose_f16_kernel<<<(4 * DD * DD + T - 1) / T, T, 0, stream>>>(W_mlp2,  Wt_mlp2,  4 * DD, DD);

    dim3 gblk(32, 8, 1);   // 8 waves; 128(M) x 64(N) per block

    // ---- 1) LN + modulate + path gather -> tp (f16) ----
    ln_mod_kernel<<<MTOT, 128, 0, stream>>>(x, t, paths, tp_h, 1);

    // ---- 2) xz = tp @ W_in : 4096 x 1024 x 512 ----
    wmma_gemm_kernel<<<dim3((2 * DD) / 64, MTOT / 128), gblk, 0, stream>>>(
        tp_h, Wt_in, nullptr, xz, nullptr, MTOT, 2 * DD, DD, DD, 0);

    // ---- 3) depthwise conv + SiLU ----
    conv_silu_kernel<<<(MTOT * DD + T - 1) / T, T, 0, stream>>>(xz, conv_w, conv_b, xc, xc_h);

    // ---- 4) x_dbl = xc @ W_xproj : 4096 x 64 x 512 (f32 + f16) ----
    wmma_gemm_kernel<<<dim3(64 / 64, MTOT / 128), gblk, 0, stream>>>(
        xc_h, Wt_xproj, nullptr, x_dbl, x_dbl_h, MTOT, 64, DD, DD, 4);

    // ---- 5) dt = softplus(x_dbl[:, :32] @ W_dt + b_dt) : 4096 x 512 x 32 ----
    wmma_gemm_kernel<<<dim3(DD / 64, MTOT / 128), gblk, 0, stream>>>(
        x_dbl_h, Wt_dt, b_dt, dt_buf, nullptr, MTOT, DD, DTR_, 64, 2);

    // ---- 6) selective scan + fused gate -> ym (f16) ----
    ssm_scan_kernel<<<NSEQ, DD, 0, stream>>>(dt_buf, x_dbl, xz, xc, A_log, D_param, ym_h);

    // ---- 7) mo = ym @ W_out : 4096 x 512 x 512 ----
    wmma_gemm_kernel<<<dim3(DD / 64, MTOT / 128), gblk, 0, stream>>>(
        ym_h, Wt_out, nullptr, mo, nullptr, MTOT, DD, DD, DD, 0);

    // ---- 8) un-permute + mean over paths + residuals -> x1 ----
    mean_residual_kernel<<<(MBL * DD + T - 1) / T, T, 0, stream>>>(mo, prev, x, y, t, x1);

    // ---- 9) LN + modulate (mlp) -> f16 ----
    ln_mod_kernel<<<MBL, 128, 0, stream>>>(x1, t, nullptr, hmod_h, 0);

    // ---- 10) mlp1 = gelu(h @ W_mlp1 + b1) : 512 x 2048 x 512 -> f16 ----
    wmma_gemm_kernel<<<dim3((4 * DD) / 64, MBL / 128), gblk, 0, stream>>>(
        hmod_h, Wt_mlp1, b_mlp1, nullptr, mlp1_h, MBL, 4 * DD, DD, DD, 3);

    // ---- 11) mlp2 = mlp1 @ W_mlp2 + b2 : 512 x 512 x 2048 ----
    wmma_gemm_kernel<<<dim3(DD / 64, MBL / 128), gblk, 0, stream>>>(
        mlp1_h, Wt_mlp2, b_mlp2, mlp2, nullptr, MBL, DD, 4 * DD, 4 * DD, 1);

    // ---- 12) out = x1 + g_mlp * mlp2 ----
    final_residual_kernel<<<(MBL * DD + T - 1) / T, T, 0, stream>>>(x1, mlp2, t, out);
}
